// OnlineFlashMTPModel_17532056502648
// MI455X (gfx1250) — matrix operand order repaired and checked by reading, not compiled, not run on hardware
//
#include <hip/hip_runtime.h>
#include <hip/hip_bf16.h>

// ---------------- problem constants (from reference) ----------------
#define BB      2
#define SS      4096
#define DD      2048
#define VV      32000
#define HH      16
#define HDm     128
#define BSq     16
#define NAn     128
#define MASKID  31999
#define DFFn    8192
#define NPn     (SS - BSq + 1)      // 4081
#define ROWS_X  (BB * NAn * BSq)    // 4096 query rows
#define ROWS_KV (BB * NAn * 17)     // 4352 kv rows (ctx + 16 queries)
#define CHk     512                 // lm-head row chunk

typedef __attribute__((ext_vector_type(16))) __bf16          v16bf;
typedef __attribute__((ext_vector_type(8)))  float           v8f;
typedef __attribute__((ext_vector_type(4)))  unsigned short  us4;
typedef __attribute__((ext_vector_type(8)))  unsigned short  us8;

__device__ __forceinline__ unsigned short f2bf(float f) {
    union { float f; unsigned u; } x; x.f = f;
    unsigned r = x.u + 0x7fffu + ((x.u >> 16) & 1u);   // RNE
    return (unsigned short)(r >> 16);
}

// ---------------- weight conversion: f32 [K][N] -> bf16 [N][K] (tiled transpose) --------
__global__ __launch_bounds__(256) void k_w2bf_t(const float* __restrict__ W,
                                                unsigned short* __restrict__ WT,
                                                int K, int N) {
    __shared__ unsigned short tile[32][33];
    int k0 = blockIdx.y * 32, n0 = blockIdx.x * 32;
#pragma unroll
    for (int j = 0; j < 4; j++) {
        int i = threadIdx.x + j * 256;
        int r = i >> 5, c = i & 31;                    // r: k, c: n (coalesced read)
        tile[c][r] = f2bf(W[(long)(k0 + r) * N + n0 + c]);
    }
    __syncthreads();
#pragma unroll
    for (int j = 0; j < 4; j++) {
        int i = threadIdx.x + j * 256;
        int r = i >> 5, c = i & 31;                    // r: n, c: k (coalesced write)
        WT[(long)(n0 + r) * K + k0 + c] = tile[r][c];
    }
}

// f32 [N][K] -> bf16 [N][K] straight (lm_head is already N-major)
__global__ void k_w2bf(const float* __restrict__ W, unsigned short* __restrict__ WT) {
    long idx = ((long)blockIdx.x * 256 + threadIdx.x) * 4;
    float4 fv = *(const float4*)(W + idx);
    us4 pk = {f2bf(fv.x), f2bf(fv.y), f2bf(fv.z), f2bf(fv.w)};
    *(us4*)(WT + idx) = pk;
}

// ---------------- anchor selection (rank-select 128 smallest rv) ----------------
__global__ __launch_bounds__(256) void k_anchors(const int* ids, const float* lm,
                                                 const float* rv, int* anchors,
                                                 float* keep, int* ftok) {
    int b = blockIdx.x;
    __shared__ float srv[NPn];
    __shared__ int   sel[NAn];
    __shared__ int   ssort[NAn];
    __shared__ int   svc;
    if (threadIdx.x == 0) svc = 0;
    __syncthreads();
    int lvc = 0;
    for (int i = threadIdx.x; i < NPn; i += 256) {
        bool v = lm[b * SS + i] > 0.5f;
        srv[i] = v ? rv[b * NPn + i] : 2.0f;
        if (v) lvc++;
    }
    atomicAdd(&svc, lvc);
    __syncthreads();
    for (int i = threadIdx.x; i < NPn; i += 256) {
        float ri = srv[i];
        int rank = 0;
        for (int j = 0; j < NPn; j++) {
            float rj = srv[j];
            rank += (rj < ri) || (rj == ri && j < i);
        }
        if (rank < NAn) sel[rank] = (ri < 1.5f) ? i : (SS + 1);
    }
    __syncthreads();
    if (threadIdx.x < NAn) {
        int t = threadIdx.x, v = sel[t], r = 0;
        for (int j = 0; j < NAn; j++) { int u = sel[j]; r += (u < v) || (u == v && j < t); }
        ssort[r] = v;
    }
    __syncthreads();
    if (threadIdx.x < NAn) {
        int t = threadIdx.x;
        int lim = svc < NAn ? svc : NAn;
        int kp = (t < lim) ? 1 : 0;
        int a = kp ? ssort[t] : 0;
        anchors[b * NAn + t] = a;
        keep[b * NAn + t] = kp ? 1.f : 0.f;
        int ca = a < 0 ? 0 : (a > SS - 1 ? SS - 1 : a);
        ftok[b * NAn + t] = kp ? ids[b * SS + ca] : MASKID;
    }
}

// ---------------- embedding / ctx gather ----------------
__global__ void k_embed(const float* emb, const int* ftok, float* X) {
    long idx = (long)blockIdx.x * 256 + threadIdx.x;   // ROWS_X * DD
    int c = (int)(idx % DD);
    long row = idx / DD;
    int qi = (int)(row % BSq);
    long bn = row / BSq;
    int tok = (qi == 0) ? ftok[bn] : MASKID;
    X[idx] = emb[(long)tok * DD + c];
}

__global__ void k_ctx(const float* hidden, const int* anchors, float* CTX) {
    long idx = (long)blockIdx.x * 256 + threadIdx.x;   // BB*NAn*DD
    int c = (int)(idx % DD);
    long bn = idx / DD;
    int b = (int)(bn / NAn);
    int a = anchors[bn];
    int cp = a - 1; if (cp < 0) cp = 0;
    CTX[idx] = hidden[((long)b * SS + cp) * DD + c];
}

// ---------------- RMSNorm rows ----------------
__global__ __launch_bounds__(256) void k_rms(const float* in, const float* w,
                                             float* out, int Dd) {
    long row = blockIdx.x;
    const float* src = in + row * Dd;
    __shared__ float red[256];
    float s = 0.f;
    for (int c = threadIdx.x; c < Dd; c += 256) { float v = src[c]; s += v * v; }
    red[threadIdx.x] = s; __syncthreads();
    for (int st = 128; st > 0; st >>= 1) {
        if (threadIdx.x < st) red[threadIdx.x] += red[threadIdx.x + st];
        __syncthreads();
    }
    float inv = rsqrtf(red[0] / (float)Dd + 1e-6f);
    for (int c = threadIdx.x; c < Dd; c += 256) out[row * Dd + c] = src[c] * inv * w[c];
}

// kv_in rows: j==0 -> rms(ctx, norm_chs); j>0 -> rms(x, norm_attn)
__global__ __launch_bounds__(256) void k_kvin(const float* X, const float* CTX,
                                              const float* wa, const float* wc, float* KV) {
    long row = blockIdx.x;            // ROWS_KV
    int j = (int)(row % 17);
    long bn = row / 17;
    const float* src; const float* w;
    if (j == 0) { src = CTX + bn * DD; w = wc; }
    else        { src = X + (bn * BSq + (j - 1)) * DD; w = wa; }
    __shared__ float red[256];
    float s = 0.f;
    for (int c = threadIdx.x; c < DD; c += 256) { float v = src[c]; s += v * v; }
    red[threadIdx.x] = s; __syncthreads();
    for (int st = 128; st > 0; st >>= 1) {
        if (threadIdx.x < st) red[threadIdx.x] += red[threadIdx.x + st];
        __syncthreads();
    }
    float inv = rsqrtf(red[0] / (float)DD + 1e-6f);
    for (int c = threadIdx.x; c < DD; c += 256) KV[row * DD + c] = src[c] * inv * w[c];
}

// ---------------- bf16 WMMA GEMM: C[M,N] = A[M,K](f32) @ Bt[N,K](bf16)^T ----------------
// REQUIRES: M % 64 == 0, N % 128 == 0, K % 32 == 0 (all call sites comply).
// LDS tiles stored in WMMA fragment order (32B per lane per subtile).
// Register-level pipeline: store tile k -> LDS, sync, issue global loads for
// tile k+1, run 4 WMMAs on tile k, sync.
__global__ __launch_bounds__(256) void k_gemm(const float* __restrict__ A,
                                              const unsigned short* __restrict__ Bt,
                                              float* __restrict__ C,
                                              int M, int N, int K) {
    __shared__ alignas(32) unsigned short Af[4 * 32 * 16];   // [mSub][lane][e]  4KB
    __shared__ alignas(32) unsigned short Bf[8 * 32 * 16];   // [nSub][lane][e]  8KB
    int m0 = blockIdx.y * 64, n0 = blockIdx.x * 128;
    int tid  = threadIdx.x;
    int w    = tid >> 5, lane = tid & 31;
    int wm   = (w & 3) * 16;          // wave's m-subtile offset
    int wnS  = (w >> 2) * 4;          // wave's first n-subtile (4 subtiles/wave)

    // per-thread staging geometry (A: 2 x float4; B: 2 x 8-bf16)
    int aIdxL[2]; const float* aPtr[2];
    int bIdxL[2]; const unsigned short* bPtr[2];
#pragma unroll
    for (int j = 0; j < 2; j++) {
        int i  = tid + j * 256;            // 0..511
        int r  = i >> 3;                   // 0..63
        int c4 = (i & 7) * 4;              // 0,4,...,28
        int half = (c4 >> 3) & 1;
        int t2   = c4 - 8 * half;
        int e    = (t2 < 8) ? t2 : t2 - 8;
        aIdxL[j] = ((((r >> 4) << 5) + (half << 4) + (r & 15)) << 4) + e;
        aPtr[j]  = A + (long)(m0 + r) * K + c4;

        int cc = i >> 2;                   // 0..127
        int k8 = (i & 3) * 8;              // 0,8,16,24
        int sub   = cc >> 4;
        int lane_ = (cc & 15) + (((k8 >> 3) & 1) << 4);
        bIdxL[j] = ((((sub << 5) + lane_) << 4)) + ((k8 & 16) >> 1);
        bPtr[j]  = Bt + (long)(n0 + cc) * K + k8;
    }

    v8f acc[4] = {v8f{}, v8f{}, v8f{}, v8f{}};
    float4 ar[2]; us8 br[2];
#pragma unroll
    for (int j = 0; j < 2; j++) {          // prefetch tile 0
        ar[j] = *(const float4*)(aPtr[j]);
        br[j] = *(const us8*)(bPtr[j]);
    }

    for (int k0 = 0; k0 < K; k0 += 32) {
        // ---- commit prefetched tile to LDS (fragment order) ----
#pragma unroll
        for (int j = 0; j < 2; j++) {
            us4 pk = {f2bf(ar[j].x), f2bf(ar[j].y), f2bf(ar[j].z), f2bf(ar[j].w)};
            *(us4*)&Af[aIdxL[j]] = pk;
            *(us8*)&Bf[bIdxL[j]] = br[j];
        }
        __syncthreads();
        // ---- issue next tile's global loads (overlaps WMMA below) ----
        if (k0 + 32 < K) {
#pragma unroll
            for (int j = 0; j < 2; j++) {
                ar[j] = *(const float4*)(aPtr[j] + k0 + 32);
                br[j] = *(const us8*)(bPtr[j] + k0 + 32);
            }
        }
        // ---- compute: 1 A-frag reused over 4 B-frags -> 4 WMMAs ----
        v16bf av = *(const v16bf*)&Af[(((wm >> 4) << 5) + lane) << 4];
#pragma unroll
        for (int s = 0; s < 4; s++) {
            v16bf bv = *(const v16bf*)&Bf[(((wnS + s) << 5) + lane) << 4];
            acc[s] = __builtin_amdgcn_wmma_f32_16x16x32_bf16(false, av, false, bv,
                                                             (short)0, acc[s], false, false);
        }
        __syncthreads();
    }
    // ---- store C (16x16 f32 D layout: VGPR r -> M = r + 8*half, lane -> N) ----
    int half = lane >> 4, l16 = lane & 15;
#pragma unroll
    for (int s = 0; s < 4; s++) {
        int gn = n0 + ((wnS + s) << 4) + l16;
#pragma unroll
        for (int r = 0; r < 8; r++) {
            int gm = m0 + wm + r + 8 * half;
            C[(long)gm * N + gn] = acc[s][r];
        }
    }
}

// ---------------- RoPE (in place) ----------------
__global__ void k_rope(float* buf, const int* anchors, long total, int isQ) {
    long idx = (long)blockIdx.x * 256 + threadIdx.x;
    if (idx >= total) return;
    int i = (int)(idx & 63);
    int h = (int)((idx >> 6) & 15);
    long row = idx >> 10;
    int j = (int)(row % 17);
    long bn = row / 17;
    if (isQ && j == 0) return;
    int a = anchors[bn];
    int pos = (j == 0) ? (a - 1 > 0 ? a - 1 : 0) : (a + j - 1);
    long base = row * DD + (long)h * HDm;
    float x1 = buf[base + i], x2 = buf[base + 64 + i];
    float inv = __expf(-(2.f * i / (float)HDm) * 9.210340371976184f);  // ln(1e4)
    float ang = (float)pos * inv;
    float cs = __cosf(ang), sn = __sinf(ang);
    buf[base + i]      = x1 * cs - x2 * sn;
    buf[base + 64 + i] = x2 * cs + x1 * sn;
}

// ---------------- tiny flash attention per (b, anchor, head) ----------------
__global__ __launch_bounds__(128) void k_attn(const float* Q, const float* K,
                                              const float* V, float* AO) {
    int unit = blockIdx.x;
    int h = unit & 15;
    long bn = unit >> 4;
    __shared__ float qs[16 * 128], ks[17 * 128], vs[17 * 128], sc[16 * 17];
    int t = threadIdx.x;
    long qbase = (bn * 17 + 1) * DD + (long)h * HDm;
    long kbase = (bn * 17) * DD + (long)h * HDm;
    for (int i = t; i < 16 * 128; i += 128) {
        int r = i >> 7, c = i & 127;
        qs[i] = Q[qbase + (long)r * DD + c];
    }
    for (int i = t; i < 17 * 128; i += 128) {
        int r = i / 128, c = i % 128;
        ks[i] = K[kbase + (long)r * DD + c];
        vs[i] = V[kbase + (long)r * DD + c];
    }
    __syncthreads();
    const float scale = 0.08838834764831845f;      // 1/sqrt(128)
    for (int p = t; p < 16 * 17; p += 128) {
        int qi = p / 17, kj = p % 17;
        float s = 0.f;
        for (int d = 0; d < 128; d++) s += qs[qi * 128 + d] * ks[kj * 128 + d];
        sc[p] = s * scale;
    }
    __syncthreads();
    if (t < 16) {
        float mx = -1e30f;
        for (int kj = 0; kj < 17; kj++) mx = fmaxf(mx, sc[t * 17 + kj]);
        float sum = 0.f;
        for (int kj = 0; kj < 17; kj++) { float e = __expf(sc[t * 17 + kj] - mx); sc[t * 17 + kj] = e; sum += e; }
        float inv = 1.f / sum;
        for (int kj = 0; kj < 17; kj++) sc[t * 17 + kj] *= inv;
    }
    __syncthreads();
    for (int i = t; i < 16 * 128; i += 128) {
        int qi = i >> 7, d = i & 127;
        float o = 0.f;
        for (int kj = 0; kj < 17; kj++) o += sc[qi * 17 + kj] * vs[kj * 128 + d];
        AO[(bn * 16 + qi) * DD + (long)h * HDm + d] = o;
    }
}

// ---------------- residual add (optionally gated by keep) ----------------
__global__ void k_resid(float* X, const float* Yv, const float* keep, long n) {
    long idx = (long)blockIdx.x * 256 + threadIdx.x;
    if (idx >= n) return;
    float kf = keep ? keep[(idx / DD) / BSq] : 1.f;
    X[idx] += Yv[idx] * kf;
}

// ---------------- silu(g) * u -> g ----------------
__global__ void k_silu(float* G, const float* U, long n) {
    long idx = (long)blockIdx.x * 256 + threadIdx.x;
    if (idx >= n) return;
    float g = G[idx];
    G[idx] = (g / (1.f + __expf(-g))) * U[idx];
}

// ---------------- per-row CE + argmax over V=32000 ----------------
__global__ __launch_bounds__(256) void k_loss(const float* LOG, const int* ids,
                                              const float* lmask, const int* anchors,
                                              const float* keep, int baseRow, float* scal) {
    int r = blockIdx.x;
    int row = baseRow + r;
    int qi = row % BSq;
    int bn = row / BSq;
    int b = bn / NAn;
    const float* lr = LOG + (long)r * VV;
    __shared__ float smax[256]; __shared__ int sidx[256]; __shared__ float ssum[256];
    float mx = -1e30f; int mi = 0;
    for (int c = threadIdx.x; c < VV; c += 256) {
        float v = lr[c];
        if (v > mx || (v == mx && c < mi)) { mx = v; mi = c; }
    }
    smax[threadIdx.x] = mx; sidx[threadIdx.x] = mi; __syncthreads();
    for (int st = 128; st > 0; st >>= 1) {
        if (threadIdx.x < st) {
            float o = smax[threadIdx.x + st]; int oi = sidx[threadIdx.x + st];
            if (o > smax[threadIdx.x] || (o == smax[threadIdx.x] && oi < sidx[threadIdx.x])) {
                smax[threadIdx.x] = o; sidx[threadIdx.x] = oi;
            }
        }
        __syncthreads();
    }
    float rowmax = smax[0]; int pred = sidx[0];
    float s = 0.f;
    for (int c = threadIdx.x; c < VV; c += 256) s += __expf(lr[c] - rowmax);
    ssum[threadIdx.x] = s; __syncthreads();
    for (int st = 128; st > 0; st >>= 1) {
        if (threadIdx.x < st) ssum[threadIdx.x] += ssum[threadIdx.x + st];
        __syncthreads();
    }
    if (threadIdx.x == 0) {
        int a = anchors[bn];
        int label = a + qi;
        int valid = label < SS;
        int safe = valid ? label : SS - 1;
        int target = ids[b * SS + safe];
        float wgt = keep[bn] * (valid ? 1.f : 0.f) * ((qi > 0) ? 1.f : 0.f) * lmask[b * SS + safe];
        float logp = lr[target] - rowmax - __logf(ssum[0]);
        atomicAdd(&scal[0], -logp * wgt);
        atomicAdd(&scal[1], wgt);
        if (wgt > 0.5f && pred == target) atomicAdd(&scal[2], 1.f);
    }
}

__global__ void k_final(const float* scal, float* out) {
    if (threadIdx.x == 0) {
        out[0] = scal[0] / (scal[1] + 1e-6f);
        out[1] = scal[2] / (scal[1] + 1e-6f);
    }
}

// ---------------- host orchestration ----------------
extern "C" void kernel_launch(void* const* d_in, const int* in_sizes, int n_in,
                              void* d_out, int out_size, void* d_ws, size_t ws_size,
                              hipStream_t stream) {
    const int*   input_ids = (const int*)  d_in[0];
    const float* hidden    = (const float*)d_in[1];
    const float* loss_mask = (const float*)d_in[2];
    const float* rand_vals = (const float*)d_in[3];
    const float* embed     = (const float*)d_in[4];
    const float* lm_head   = (const float*)d_in[5];
    const float* wq        = (const float*)d_in[6];
    const float* wk        = (const float*)d_in[7];
    const float* wv        = (const float*)d_in[8];
    const float* wo        = (const float*)d_in[9];
    const float* w_gate    = (const float*)d_in[10];
    const float* w_up      = (const float*)d_in[11];
    const float* w_down    = (const float*)d_in[12];
    const float* norm_attn = (const float*)d_in[13];
    const float* norm_mlp  = (const float*)d_in[14];
    const float* norm_out  = (const float*)d_in[15];
    const float* norm_chs  = (const float*)d_in[16];

    char* basep = (char*)d_ws;
    size_t off = 0;
    auto alloc = [&](size_t bytes) -> char* {
        char* p = basep + off;
        off = (off + bytes + 255) & ~(size_t)255;
        return p;
    };
    int*   anchors = (int*)  alloc((size_t)BB * NAn * 4);
    float* keep    = (float*)alloc((size_t)BB * NAn * 4);
    int*   ftok    = (int*)  alloc((size_t)BB * NAn * 4);
    float* scal    = (float*)alloc(4 * sizeof(float));
    float* X   = (float*)alloc((size_t)ROWS_X  * DD  * 4);
    float* CTX = (float*)alloc((size_t)BB * NAn * DD * 4);
    float* KV  = (float*)alloc((size_t)ROWS_KV * DD  * 4);
    float* Qb  = (float*)alloc((size_t)ROWS_KV * DD  * 4);
    float* Kb  = (float*)alloc((size_t)ROWS_KV * DD  * 4);
    float* Vb  = (float*)alloc((size_t)ROWS_KV * DD  * 4);
    float* AO  = (float*)alloc((size_t)ROWS_X  * DD  * 4);
    float* O2  = (float*)alloc((size_t)ROWS_X  * DD  * 4);
    float* Hb  = (float*)alloc((size_t)ROWS_X  * DD  * 4);
    float* G   = (float*)alloc((size_t)ROWS_X  * DFFn * 4);
    float* U   = (float*)alloc((size_t)ROWS_X  * DFFn * 4);
    float* LOG = (float*)alloc((size_t)CHk * VV * 4);
    // bf16 transposed weights [N][K]
    unsigned short* WqT = (unsigned short*)alloc((size_t)DD * DD * 2);
    unsigned short* WkT = (unsigned short*)alloc((size_t)DD * DD * 2);
    unsigned short* WvT = (unsigned short*)alloc((size_t)DD * DD * 2);
    unsigned short* WoT = (unsigned short*)alloc((size_t)DD * DD * 2);
    unsigned short* WgT = (unsigned short*)alloc((size_t)DFFn * DD * 2);
    unsigned short* WuT = (unsigned short*)alloc((size_t)DFFn * DD * 2);
    unsigned short* WdT = (unsigned short*)alloc((size_t)DD * DFFn * 2);
    unsigned short* WlT = (unsigned short*)alloc((size_t)VV * DD * 2);
    (void)ws_size; (void)in_sizes; (void)n_in; (void)out_size;

    hipMemsetAsync(scal, 0, 4 * sizeof(float), stream);

    // one-time (per launch) weight bf16 conversion / transpose
    k_w2bf_t<<<dim3(DD / 32, DD / 32), 256, 0, stream>>>(wq, WqT, DD, DD);
    k_w2bf_t<<<dim3(DD / 32, DD / 32), 256, 0, stream>>>(wk, WkT, DD, DD);
    k_w2bf_t<<<dim3(DD / 32, DD / 32), 256, 0, stream>>>(wv, WvT, DD, DD);
    k_w2bf_t<<<dim3(DD / 32, DD / 32), 256, 0, stream>>>(wo, WoT, DD, DD);
    k_w2bf_t<<<dim3(DFFn / 32, DD / 32), 256, 0, stream>>>(w_gate, WgT, DD, DFFn);
    k_w2bf_t<<<dim3(DFFn / 32, DD / 32), 256, 0, stream>>>(w_up,   WuT, DD, DFFn);
    k_w2bf_t<<<dim3(DD / 32, DFFn / 32), 256, 0, stream>>>(w_down, WdT, DFFn, DD);
    k_w2bf<<<(unsigned)(((size_t)VV * DD) / 4 / 256), 256, 0, stream>>>(lm_head, WlT);

    k_anchors<<<BB, 256, 0, stream>>>(input_ids, loss_mask, rand_vals, anchors, keep, ftok);
    k_embed<<<(ROWS_X * DD) / 256, 256, 0, stream>>>(embed, ftok, X);
    k_ctx<<<(BB * NAn * DD) / 256, 256, 0, stream>>>(hidden, anchors, CTX);
    k_kvin<<<ROWS_KV, 256, 0, stream>>>(X, CTX, norm_attn, norm_chs, KV);

    dim3 blk(256);
    k_gemm<<<dim3(DD / 128, ROWS_KV / 64), blk, 0, stream>>>(KV, WqT, Qb, ROWS_KV, DD, DD);
    k_gemm<<<dim3(DD / 128, ROWS_KV / 64), blk, 0, stream>>>(KV, WkT, Kb, ROWS_KV, DD, DD);
    k_gemm<<<dim3(DD / 128, ROWS_KV / 64), blk, 0, stream>>>(KV, WvT, Vb, ROWS_KV, DD, DD);

    long ropeN = (long)ROWS_KV * HH * 64;
    k_rope<<<(unsigned)((ropeN + 255) / 256), 256, 0, stream>>>(Qb, anchors, ropeN, 1);
    k_rope<<<(unsigned)((ropeN + 255) / 256), 256, 0, stream>>>(Kb, anchors, ropeN, 0);

    k_attn<<<BB * NAn * HH, 128, 0, stream>>>(Qb, Kb, Vb, AO);
    k_gemm<<<dim3(DD / 128, ROWS_X / 64), blk, 0, stream>>>(AO, WoT, O2, ROWS_X, DD, DD);
    k_resid<<<(ROWS_X * DD) / 256, 256, 0, stream>>>(X, O2, keep, (long)ROWS_X * DD);

    k_rms<<<ROWS_X, 256, 0, stream>>>(X, norm_mlp, Hb, DD);
    k_gemm<<<dim3(DFFn / 128, ROWS_X / 64), blk, 0, stream>>>(Hb, WgT, G, ROWS_X, DFFn, DD);
    k_gemm<<<dim3(DFFn / 128, ROWS_X / 64), blk, 0, stream>>>(Hb, WuT, U, ROWS_X, DFFn, DD);
    k_silu<<<(unsigned)(((long)ROWS_X * DFFn) / 256), 256, 0, stream>>>(G, U, (long)ROWS_X * DFFn);
    k_gemm<<<dim3(DD / 128, ROWS_X / 64), blk, 0, stream>>>(G, WdT, O2, ROWS_X, DD, DFFn);
    k_resid<<<(ROWS_X * DD) / 256, 256, 0, stream>>>(X, O2, nullptr, (long)ROWS_X * DD);

    k_rms<<<ROWS_X, 256, 0, stream>>>(X, norm_out, Hb, DD);

    for (int c = 0; c < ROWS_X / CHk; c++) {
        k_gemm<<<dim3(VV / 128, CHk / 64), blk, 0, stream>>>(
            Hb + (size_t)c * CHk * DD, WlT, LOG, CHk, VV, DD);
        k_loss<<<CHk, 256, 0, stream>>>(LOG, input_ids, loss_mask, anchors, keep, c * CHk, scal);
    }
    k_final<<<1, 32, 0, stream>>>(scal, (float*)d_out);
}
